// unit_gtcn_C_9921374454163
// MI455X (gfx1250) — matrix operand order repaired and checked by reading, not compiled, not run on hardware
//
#include <hip/hip_runtime.h>
#include <hip/hip_bf16.h>
#include <math.h>

typedef _Float16 v16h __attribute__((ext_vector_type(16)));
typedef _Float16 v8h  __attribute__((ext_vector_type(8)));
typedef float    v8f  __attribute__((ext_vector_type(8)));

#define kN   64
#define kC   64
#define kT   300
#define kV   25
#define kIC  16
#define kTV  (kT * kV)        // 7500
#define kXP  7504             // padded T*V row stride (16B-aligned rows)
#define kCT  (kC * kT)        // 19200
#define kKA  (kIC * kT)       // 4800 (attention K)
#define kTVTiles 469          // ceil(7500/16)
#define kTotal ((size_t)kN * kC * kT * kV)   // 30,720,000

union F16x16 { v16h v; _Float16 h[16]; };
union F32x8  { v8f  v; float    f[8]; };

// A-fragment K index for element e (0..15), lane-half h (ISA 7.12.2, 16-bit A 16x32)
__device__ __forceinline__ int a_kmap(int e, int h) {
    return (e < 8) ? (8 * h + e) : (16 + 8 * h + (e - 8));
}

// Two aligned b128 loads: 16 contiguous f16
__device__ __forceinline__ void load_row16(F16x16& d, const _Float16* p) {
    *(v8h*)(&d.h[0]) = *(const v8h*)(p);
    *(v8h*)(&d.h[8]) = *(const v8h*)(p + 8);
}
// A-fragment from a K-contiguous row: h[0..7]=row[8h..], h[8..15]=row[16+8h..]
__device__ __forceinline__ void load_afrag(F16x16& d, const _Float16* row, int half) {
    *(v8h*)(&d.h[0]) = *(const v8h*)(row + 8 * half);
    *(v8h*)(&d.h[8]) = *(const v8h*)(row + 16 + 8 * half);
}

// -------------------------------------------------- pack f32 weights into WMMA A-fragment tables
// wp[((mt*nks + ks)*32 + lane)*16 + e] = (f16) w[(mt*16 + lane&15)*Ktot + ks*32 + a_kmap(e, lane>>4)]
__global__ void k_wpack(const float* __restrict__ w, _Float16* __restrict__ wp,
                        int Ktot, int mtiles) {
    int nks = Ktot / 32;
    int total = mtiles * nks * 512;
    int idx = blockIdx.x * blockDim.x + threadIdx.x;
    if (idx >= total) return;
    int e    = idx & 15;
    int lane = (idx >> 4) & 31;
    int ks   = (idx >> 9) % nks;
    int mt   = idx / (nks * 512);
    int half = lane >> 4, l15 = lane & 15;
    int k = ks * 32 + a_kmap(e, half);
    wp[idx] = (_Float16)w[(mt * 16 + l15) * Ktot + k];
}

// -------------------------------------------------- cast x -> f16 (two layouts)
// xh : (n, c, kXP)   channel rows, 16B-aligned stride
// xct: (n, c*T, 32)  x_flat rows padded to 32, cols 25..31 zeroed (clean K pad)
__global__ void k_cast(const float* __restrict__ src, _Float16* __restrict__ xh,
                       _Float16* __restrict__ xct) {
    size_t i = (size_t)blockIdx.x * blockDim.x + threadIdx.x;
    if (i >= kTotal) return;
    _Float16 h = (_Float16)src[i];
    int v = (int)(i % kV);
    size_t r = i / kV;
    int t = (int)(r % kT);
    size_t r2 = r / kT;
    int c = (int)(r2 % kC);
    int n = (int)(r2 / kC);
    xh[((size_t)n * kC + c) * kXP + t * kV + v] = h;
    size_t row = (size_t)n * kCT + c * kT + t;
    xct[row * 32 + v] = h;
    if (v < 7) xct[row * 32 + 25 + v] = (_Float16)0.f;   // zero K-pad columns
}

// -------------------------------------------------- A_mix (NS,25,25)
__global__ void k_amix(const float* __restrict__ A, const float* __restrict__ w,
                       float* __restrict__ amix) {
    __shared__ float c4[kV * kV];
    int s = blockIdx.x, col = threadIdx.x;
    const float* As = A + s * kV * kV;
    if (col < kV) {
        for (int r = 0; r < kV; ++r) {
            float a = As[r * kV + col];
            float a2 = a * a;
            float eye = (r == col) ? 1.f : 0.f;
            c4[r * kV + col] = (8.f * a2 * a2 - 8.f * a2 + eye) * (1.f / (float)kV);
        }
    }
    __syncthreads();
    if (col < kV) {
        float m = -1e30f;
        for (int r = 0; r < kV; ++r) m = fmaxf(m, c4[r * kV + col]);
        float sum = 0.f;
        for (int r = 0; r < kV; ++r) sum += __expf(c4[r * kV + col] - m);
        float inv = 1.f / sum;
        for (int r = 0; r < kV; ++r) {
            float a = As[r * kV + col];
            float a2 = a * a, a3 = a2 * a, a4 = a2 * a2;
            float eye = (r == col) ? 1.f : 0.f;
            float ch4 = 8.f * a4 - 8.f * a2 + eye;
            float ch3 = 4.f * a3 - 3.f * a;
            float ch2 = 2.f * a2 - eye;
            float sm  = __expf(c4[r * kV + col] - m) * inv;
            amix[s * kV * kV + r * kV + col] =
                w[0] * a + w[1] * sm + w[2] * ch4 + w[3] * ch3 + w[4] * ch2;
        }
    }
}

// -------------------------------------------------- feature store (compile-time layout)
// TRANSPOSED==1 -> f1t layout (n, 32, K=IC*T): row v, col ic*T+t
// TRANSPOSED==0 -> f2p layout (n, IC*T, 32):  row ic*T+t, col v
template <int TRANSPOSED>
__device__ __forceinline__ void store_feat(_Float16* __restrict__ f, int n, int half,
                                           int l15, int tvbase, const F32x8& o,
                                           const float* __restrict__ bias) {
    int tv = tvbase + l15;
    if (tv >= kTV) return;
    int t = tv / kV, v = tv - t * kV;
    if (TRANSPOSED) {
        _Float16* base = f + ((size_t)n * 32 + v) * kKA + t;
        #pragma unroll
        for (int r = 0; r < 8; ++r) {
            int M = r + 8 * half;
            base[(size_t)M * kT] = (_Float16)(o.f[r] + bias[M]);
        }
    } else {
        _Float16* base = f + ((size_t)n * kKA + t) * 32 + v;
        #pragma unroll
        for (int r = 0; r < 8; ++r) {
            int M = r + 8 * half;
            base[(size_t)M * kT * 32] = (_Float16)(o.f[r] + bias[M]);
        }
    }
}

// -------------------------------------------------- 1x1 conv feature GEMM (M=16,K=64)
template <int TRANSPOSED>
__global__ void k_feat1x1(const _Float16* __restrict__ xh, const _Float16* __restrict__ wp,
                          const float* __restrict__ bias, _Float16* __restrict__ f) {
    int tile = blockIdx.x, n = blockIdx.y;
    int lane = threadIdx.x, half = lane >> 4, l15 = lane & 15;
    int tvbase = tile * 16;
    v8f acc = {};
    #pragma unroll
    for (int ks = 0; ks < 2; ++ks) {
        F16x16 av, bv;
        load_row16(av, wp + (size_t)(ks * 32 + lane) * 16);
        for (int e = 0; e < 16; ++e)
            bv.h[e] = xh[((size_t)n * kC + ks * 32 + a_kmap(e, half)) * kXP + tvbase + l15];
        acc = __builtin_amdgcn_wmma_f32_16x16x32_f16(false, av.v, false, bv.v,
                                                     (short)0, acc, false, false);
    }
    F32x8 o; o.v = acc;
    store_feat<TRANSPOSED>(f, n, half, l15, tvbase, o, bias);
}

// -------------------------------------------------- 9x1 temporal conv GEMM (K=576)
// time shift dt is a uniform +25*dt element shift in flattened tv; edges via clamp+select
template <int TRANSPOSED>
__global__ void k_feat9(const _Float16* __restrict__ xh, const _Float16* __restrict__ wp,
                        const float* __restrict__ bias, _Float16* __restrict__ f) {
    int tile = blockIdx.x, n = blockIdx.y;
    int lane = threadIdx.x, half = lane >> 4, l15 = lane & 15;
    int tvbase = tile * 16;
    v8f acc = {};
    for (int ks = 0; ks < 18; ++ks) {
        F16x16 av, bv;
        load_row16(av, wp + (size_t)(ks * 32 + lane) * 16);
        #pragma unroll
        for (int e = 0; e < 16; ++e) {
            int krow = ks * 32 + a_kmap(e, half);
            int c = krow / 9, dt = (krow % 9) - 4;
            const _Float16* chbase = xh + ((size_t)n * kC + c) * kXP;
            int shift = dt * kV;
            int ttv = tvbase + l15 + shift;
            int cl = ttv < 0 ? 0 : (ttv > kTV - 1 ? kTV - 1 : ttv);
            _Float16 lv = chbase[cl];                       // unconditional load
            bv.h[e] = (ttv == cl) ? lv : (_Float16)0.f;     // v_cndmask, no branch
        }
        acc = __builtin_amdgcn_wmma_f32_16x16x32_f16(false, av.v, false, bv.v,
                                                     (short)0, acc, false, false);
    }
    F32x8 o; o.v = acc;
    store_feat<TRANSPOSED>(f, n, half, l15, tvbase, o, bias);
}

// -------------------------------------------------- attention scores (M=N=32 pad, K=4800)
// f1t: (n,32,K) row-contiguous A side; f2p: (n,K,32) row-contiguous B side.
// Rows/cols >=25 compute garbage but are never stored.
__global__ void k_attn(const _Float16* __restrict__ f1t, const _Float16* __restrict__ f2p,
                       float* __restrict__ S) {
    int n = blockIdx.y;
    int mt = blockIdx.x >> 1, nt = blockIdx.x & 1;
    int lane = threadIdx.x, half = lane >> 4, l15 = lane & 15;
    int mrow = mt * 16 + l15;
    const _Float16* arow = f1t + ((size_t)n * 32 + mrow) * kKA;
    const _Float16* bbase = f2p + (size_t)n * kKA * 32 + nt * 16;
    v8f acc = {};
    for (int ks = 0; ks < 150; ++ks) {
        int kbase = ks * 32;
        F16x16 av, bv;
        load_afrag(av, arow + kbase, half);
        for (int e = 0; e < 16; ++e)
            bv.h[e] = bbase[(size_t)(kbase + a_kmap(e, half)) * 32 + l15];
        acc = __builtin_amdgcn_wmma_f32_16x16x32_f16(false, av.v, false, bv.v,
                                                     (short)0, acc, false, false);
    }
    F32x8 o; o.v = acc;
    const float scale = 1.f / (float)(kIC * kT);
    int Ncol = nt * 16 + l15;
    #pragma unroll
    for (int r = 0; r < 8; ++r) {
        int M = mt * 16 + r + 8 * half;
        if (M < kV && Ncol < kV)
            S[(size_t)n * kV * kV + M * kV + Ncol] = o.f[r] * scale;
    }
}

// -------------------------------------------------- Ai init / softmax-accum / pad
__global__ void k_ai_init(const float* __restrict__ amix_s, float* __restrict__ Ai) {
    int n = blockIdx.x, i = threadIdx.x;
    if (i < kV * kV) Ai[(size_t)n * kV * kV + i] = amix_s[i];
}

__global__ void k_softmax_acc(const float* __restrict__ S, const float* __restrict__ w,
                              int widx, float* __restrict__ Ai) {
    int n = blockIdx.x, col = threadIdx.x;
    if (col >= kV) return;
    const float* Sn = S + (size_t)n * kV * kV;
    float coef = w[widx];
    float m = -1e30f;
    for (int r = 0; r < kV; ++r) m = fmaxf(m, Sn[r * kV + col]);
    float sum = 0.f;
    for (int r = 0; r < kV; ++r) sum += __expf(Sn[r * kV + col] - m);
    float inv = 1.f / sum;
    for (int r = 0; r < kV; ++r)
        Ai[(size_t)n * kV * kV + r * kV + col] += coef * __expf(Sn[r * kV + col] - m) * inv;
}

__global__ void k_ai_pad(const float* __restrict__ Ai, _Float16* __restrict__ Aih) {
    int n = blockIdx.x, i = threadIdx.x;                 // 1024 threads: (r,c) in 32x32
    int r = i >> 5, c = i & 31;
    Aih[(size_t)n * 1024 + i] =
        (r < kV && c < kV) ? (_Float16)Ai[(size_t)n * kV * kV + r * kV + c] : (_Float16)0.f;
}

// -------------------------------------------------- z = x_flat @ Ai  (K 25->32, zero-padded)
__global__ void k_zgemm(const _Float16* __restrict__ xct, const _Float16* __restrict__ Aih,
                        _Float16* __restrict__ zh) {
    int mt = blockIdx.x, n = blockIdx.y;
    int lane = threadIdx.x, half = lane >> 4, l15 = lane & 15;
    int m = mt * 16 + l15;
    F16x16 av;
    load_afrag(av, xct + ((size_t)n * kCT + m) * 32, half);  // cols 25..31 are zeros
    const _Float16* Bn = Aih + (size_t)n * 1024;
    #pragma unroll
    for (int nt = 0; nt < 2; ++nt) {
        F16x16 bv;
        for (int e = 0; e < 16; ++e)
            bv.h[e] = Bn[a_kmap(e, half) * 32 + nt * 16 + l15];  // rows 25..31 are zeros
        v8f acc = {};
        acc = __builtin_amdgcn_wmma_f32_16x16x32_f16(false, av.v, false, bv.v,
                                                     (short)0, acc, false, false);
        F32x8 o; o.v = acc;
        int col = nt * 16 + l15;
        if (col < kV) {
            #pragma unroll
            for (int r = 0; r < 8; ++r) {
                int M = mt * 16 + r + 8 * half;
                int c = M / kT, t = M - c * kT;
                zh[((size_t)n * kC + c) * kXP + t * kV + col] = (_Float16)o.f[r];
            }
        }
    }
}

// -------------------------------------------------- y (+)= d_w @ z + d_b
__global__ void k_dgemm(const _Float16* __restrict__ zh, const _Float16* __restrict__ wp,
                        const float* __restrict__ db, float* __restrict__ y, int accflag) {
    int tile = blockIdx.x, mt = blockIdx.y, n = blockIdx.z;
    int lane = threadIdx.x, half = lane >> 4, l15 = lane & 15;
    int tvbase = tile * 16;
    v8f acc = {};
    #pragma unroll
    for (int ks = 0; ks < 2; ++ks) {
        F16x16 av, bv;
        load_row16(av, wp + (size_t)((mt * 2 + ks) * 32 + lane) * 16);
        for (int e = 0; e < 16; ++e)
            bv.h[e] = zh[((size_t)n * kC + ks * 32 + a_kmap(e, half)) * kXP + tvbase + l15];
        acc = __builtin_amdgcn_wmma_f32_16x16x32_f16(false, av.v, false, bv.v,
                                                     (short)0, acc, false, false);
    }
    F32x8 o; o.v = acc;
    int tv = tvbase + l15;
    if (tv < kTV) {
        #pragma unroll
        for (int r = 0; r < 8; ++r) {
            int M = mt * 16 + r + 8 * half;
            size_t idx = ((size_t)n * kC + M) * kTV + tv;
            float val = o.f[r] + db[M];
            y[idx] = accflag ? (y[idx] + val) : val;
        }
    }
}

// -------------------------------------------------- BN stats + final
__global__ void k_bnstats(const float* __restrict__ y, float* __restrict__ mu,
                          float* __restrict__ var) {
    int ch = blockIdx.x, tid = threadIdx.x;
    float s = 0.f, s2 = 0.f;
    for (int n = 0; n < kN; ++n) {
        const float* p = y + ((size_t)n * kC + ch) * kTV;
        for (int i = tid; i < kTV; i += 256) { float v = p[i]; s += v; s2 += v * v; }
    }
    __shared__ float sh[256], sh2[256];
    sh[tid] = s; sh2[tid] = s2; __syncthreads();
    for (int off = 128; off > 0; off >>= 1) {
        if (tid < off) { sh[tid] += sh[tid + off]; sh2[tid] += sh2[tid + off]; }
        __syncthreads();
    }
    if (tid == 0) {
        float cnt = (float)kN * (float)kTV;
        float m = sh[0] / cnt;
        mu[ch] = m;
        var[ch] = sh2[0] / cnt - m * m;
    }
}

__global__ void k_final(const float* __restrict__ y, const float* __restrict__ x,
                        const float* __restrict__ mu, const float* __restrict__ var,
                        const float* __restrict__ bnw, const float* __restrict__ bnb,
                        float* __restrict__ out) {
    size_t i = (size_t)blockIdx.x * blockDim.x + threadIdx.x;
    if (i >= kTotal) return;
    int ch = (int)((i / kTV) % kC);
    float yn = (y[i] - mu[ch]) * rsqrtf(var[ch] + 1e-5f) * bnw[ch] + bnb[ch] + x[i];
    out[i] = fmaxf(yn, 0.f);
}

// ================================================================ host orchestration
extern "C" void kernel_launch(void* const* d_in, const int* in_sizes, int n_in,
                              void* d_out, int out_size, void* d_ws, size_t ws_size,
                              hipStream_t stream) {
    (void)in_sizes; (void)n_in; (void)out_size; (void)ws_size;
    const float* x     = (const float*)d_in[0];
    const float* wts   = (const float*)d_in[1];
    const float* A     = (const float*)d_in[2];
    const float* a_w   = (const float*)d_in[3];
    const float* a_b   = (const float*)d_in[4];
    const float* b_w   = (const float*)d_in[5];
    const float* b_b   = (const float*)d_in[6];
    const float* d_w   = (const float*)d_in[7];
    const float* d_b   = (const float*)d_in[8];
    const float* t1_w  = (const float*)d_in[9];
    const float* t1_b  = (const float*)d_in[10];
    const float* t2_w  = (const float*)d_in[11];
    const float* t2_b  = (const float*)d_in[12];
    const float* st11_w1 = (const float*)d_in[13];
    const float* st11_w9 = (const float*)d_in[14];
    const float* st11_b  = (const float*)d_in[15];
    const float* st12_w1 = (const float*)d_in[16];
    const float* st12_w9 = (const float*)d_in[17];
    const float* st12_b  = (const float*)d_in[18];
    const float* bn_w  = (const float*)d_in[19];
    const float* bn_b  = (const float*)d_in[20];
    float* out = (float*)d_out;

    char* ws = (char*)d_ws;
    size_t off = 0;
    auto carve = [&](size_t bytes) -> char* {
        char* p = ws + off;
        off = (off + bytes + 255) & ~(size_t)255;
        return p;
    };
    _Float16* xh   = (_Float16*)carve((size_t)kN * kC * kXP * 2);
    _Float16* xct  = (_Float16*)carve((size_t)kN * kCT * 32 * 2);
    _Float16* f1t  = (_Float16*)carve((size_t)kN * 32 * kKA * 2);
    _Float16* f2p  = (_Float16*)carve((size_t)kN * kKA * 32 * 2);
    float*    S    = (float*)carve((size_t)kN * kV * kV * 4);
    float*    Ai   = (float*)carve((size_t)kN * kV * kV * 4);
    _Float16* Aih  = (_Float16*)carve((size_t)kN * 1024 * 2);
    _Float16* zh   = (_Float16*)carve((size_t)kN * kC * kXP * 2);
    float*    ybuf = (float*)carve(kTotal * 4);
    float*    amix = (float*)carve((size_t)3 * kV * kV * 4);
    float*    mu   = (float*)carve((size_t)kC * 4);
    float*    var  = (float*)carve((size_t)kC * 4);
    _Float16* wp1  = (_Float16*)carve((size_t)18 * 512 * 2);   // packed A-frag weights (f1 side)
    _Float16* wp2  = (_Float16*)carve((size_t)18 * 512 * 2);   // packed A-frag weights (f2 side)
    _Float16* wpD  = (_Float16*)carve((size_t)8 * 512 * 2);    // packed d_w (4 mtiles x 2 ksteps)
    (void)carve(4096);   // overread slack

    const int castBlocks = (int)((kTotal + 255) / 256);
    k_cast<<<castBlocks, 256, 0, stream>>>(x, xh, xct);
    k_amix<<<3, 32, 0, stream>>>(A, wts, amix);

    dim3 featGrid(kTVTiles, kN);
    dim3 attnGrid(4, kN);
    dim3 zGrid(kCT / 16, kN);
    dim3 dGrid(kTVTiles, kC / 16, kN);

    for (int i = 0; i < 3; ++i) {
        k_ai_init<<<kN, 640, 0, stream>>>(amix + i * kV * kV, Ai);
        // S_ab : two 1x1 feature convs (pack 2 k-steps = 1024 frag elems each)
        k_wpack<<<4, 256, 0, stream>>>(a_w + i * kIC * kC, wp1, kC, 1);
        k_wpack<<<4, 256, 0, stream>>>(b_w + i * kIC * kC, wp2, kC, 1);
        k_feat1x1<1><<<featGrid, 32, 0, stream>>>(xh, wp1, a_b + i * kIC, f1t);
        k_feat1x1<0><<<featGrid, 32, 0, stream>>>(xh, wp2, b_b + i * kIC, f2p);
        k_attn<<<attnGrid, 32, 0, stream>>>(f1t, f2p, S);
        k_softmax_acc<<<kN, 32, 0, stream>>>(S, wts, 5, Ai);
        // S_t : two 9-tap feature convs (pack 18 k-steps = 9216 frag elems each)
        k_wpack<<<36, 256, 0, stream>>>(t1_w + i * kIC * kC * 9, wp1, kC * 9, 1);
        k_wpack<<<36, 256, 0, stream>>>(t2_w + i * kIC * kC * 9, wp2, kC * 9, 1);
        k_feat9<1><<<featGrid, 32, 0, stream>>>(xh, wp1, t1_b + i * kIC, f1t);
        k_feat9<0><<<featGrid, 32, 0, stream>>>(xh, wp2, t2_b + i * kIC, f2p);
        k_attn<<<attnGrid, 32, 0, stream>>>(f1t, f2p, S);
        k_softmax_acc<<<kN, 32, 0, stream>>>(S, wts, 6, Ai);
        // S_st
        if (i == 1) {
            k_wpack<<<36, 256, 0, stream>>>(st11_w9, wp1, kC * 9, 1);
            k_wpack<<<36, 256, 0, stream>>>(st12_w9, wp2, kC * 9, 1);
            k_feat9<1><<<featGrid, 32, 0, stream>>>(xh, wp1, st11_b + 1 * kIC, f1t);
            k_feat9<0><<<featGrid, 32, 0, stream>>>(xh, wp2, st12_b + 1 * kIC, f2p);
        } else {
            int j = (i == 0) ? 0 : 1;
            k_wpack<<<4, 256, 0, stream>>>(st11_w1 + j * kIC * kC, wp1, kC, 1);
            k_wpack<<<4, 256, 0, stream>>>(st12_w1 + j * kIC * kC, wp2, kC, 1);
            k_feat1x1<1><<<featGrid, 32, 0, stream>>>(xh, wp1, st11_b + i * kIC, f1t);
            k_feat1x1<0><<<featGrid, 32, 0, stream>>>(xh, wp2, st12_b + i * kIC, f2p);
        }
        k_attn<<<attnGrid, 32, 0, stream>>>(f1t, f2p, S);
        k_softmax_acc<<<kN, 32, 0, stream>>>(S, wts, 7, Ai);

        k_ai_pad<<<kN, 1024, 0, stream>>>(Ai, Aih);
        k_zgemm<<<zGrid, 32, 0, stream>>>(xct, Aih, zh);
        k_wpack<<<16, 256, 0, stream>>>(d_w + i * kC * kC, wpD, kC, 4);
        k_dgemm<<<dGrid, 32, 0, stream>>>(zh, wpD, d_b + i * kC, ybuf, i > 0 ? 1 : 0);
    }

    k_bnstats<<<kC, 256, 0, stream>>>(ybuf, mu, var);
    k_final<<<castBlocks, 256, 0, stream>>>(ybuf, x, mu, var, bn_w, bn_b, out);
}
